// VAE_88029649699259
// MI455X (gfx1250) — matrix operand order repaired and checked by reading, not compiled, not run on hardware
//
#include <hip/hip_runtime.h>

typedef __attribute__((ext_vector_type(16))) __bf16 v16bf;
typedef __attribute__((ext_vector_type(8)))  float  v8f;
typedef unsigned short u16;
typedef unsigned int   u32;
typedef unsigned long long u64;

static constexpr int S   = 1024;
static constexpr int IN  = 2048;
static constexpr int HID = 2048;
static constexpr int DH  = 1024;

// ---------------------------------------------------------------- helpers

__device__ __forceinline__ u16 f2bf(float f) {
  u32 u = __float_as_uint(f);
  u32 r = (u + 0x7FFFu + ((u >> 16) & 1u)) >> 16;   // round-to-nearest-even
  return (u16)r;
}

union BFVec { v16bf v; uint4 q[2]; };

// A-matrix 16x32 bf16 tile, contiguous k order (same as the B loaders):
//   lane<16 : M=lane,    K = k+0..15
//   lane>=16: M=lane-16, K = k+16..31
__device__ __forceinline__ v16bf load_a_tile(const u16* W, int ldw, int row,
                                             int k, int lane) {
  int m  = lane & 15;
  int kb = k + ((lane >> 4) << 4);
  const u16* p = W + (size_t)(row + m) * (size_t)ldw + kb;
  BFVec u;
  u.q[0] = *(const uint4*)(p);        // K = kb .. kb+7
  u.q[1] = *(const uint4*)(p + 8);    // K = kb+8 .. kb+15
  return u.v;
}

// B-matrix 32x16 bf16: B[kk][n] = x[k+kk] replicated across all columns n.
//   lane holds column n=lane&15, K range = (lane>>4)*16 .. +15, contiguous.
//   (works for both global and LDS-resident x; addrspace inferred)
__device__ __forceinline__ v16bf load_b_vec(const u16* x, int k, int lane) {
  const u16* p = x + k + ((lane >> 4) << 4);
  BFVec u;
  u.q[0] = *(const uint4*)(p);
  u.q[1] = *(const uint4*)(p + 8);
  return u.v;
}

// B-matrix 32x16 bf16: B[kk][n] = W[r0+n][k+kk]  (W row-major R x K)
__device__ __forceinline__ v16bf load_b_wt(const u16* W, int ldw, int r0,
                                           int k, int lane) {
  int n = lane & 15;
  const u16* p = W + (size_t)(r0 + n) * (size_t)ldw + k + ((lane >> 4) << 4);
  BFVec u;
  u.q[0] = *(const uint4*)(p);
  u.q[1] = *(const uint4*)(p + 8);
  return u.v;
}

__device__ __forceinline__ float sigm(float x) {
  return 1.0f / (1.0f + __expf(-x));
}

// Async-stage `n` bf16 elements from global `src` into LDS at byte offset
// `lds_base`. Caller: single wave, 32 active lanes; follow with asynccnt wait.
__device__ __forceinline__ void async_stage_vec(const u16* src, u32 lds_base,
                                                int n, int lane) {
  for (int i = 0; i < n; i += 256) {            // 32 lanes x 16B = 512B/op
    u64 ga = (u64)(uintptr_t)(src + i) + (u32)(lane << 4);
    u32 la = lds_base + (u32)(i << 1) + (u32)(lane << 4);
    asm volatile("global_load_async_to_lds_b128 %0, %1, off"
                 :: "v"(la), "v"(ga) : "memory");
  }
}

// ---------------------------------------------------------------- kernels

// f32 -> bf16 conversion (one-time weight/feature cache into L2-resident bf16)
__global__ void __launch_bounds__(256)
cvt_bf16_kernel(const float* __restrict__ in, u16* __restrict__ out, int n) {
  int i = blockIdx.x * 256 + threadIdx.x;
  if (i < n) out[i] = f2bf(in[i]);
}

__global__ void __launch_bounds__(256)
zero_state_kernel(float* __restrict__ h_f, float* __restrict__ c_f,
                  u16* __restrict__ h_bf, int n) {
  int i = blockIdx.x * 256 + threadIdx.x;
  if (i < n) { h_f[i] = 0.0f; c_f[i] = 0.0f; h_bf[i] = 0; }
}

// Dense GEMM: out[T x R] = X[T x K](bf16) @ W[R x K](bf16)^T + bias[R]
// One wave per 16x16 output tile; 64-deep WMMA K-chain.
__global__ void __launch_bounds__(256)
gemm_proj_kernel(const u16* __restrict__ X, const u16* __restrict__ W,
                 const float* __restrict__ bias, float* __restrict__ out,
                 int K, int R) {
  int lane = threadIdx.x & 31;
  int wave = threadIdx.x >> 5;
  int r0 = (blockIdx.x * 8 + wave) * 16;
  int t0 = blockIdx.y * 16;
  v8f acc = {};
  for (int k = 0; k < K; k += 32) {
    v16bf a = load_a_tile(X, K, t0, k, lane);
    v16bf b = load_b_wt(W, K, r0, k, lane);
    acc = __builtin_amdgcn_wmma_f32_16x16x32_bf16(false, a, false, b,
                                                  (short)0, acc, false, false);
  }
  // D layout: VGPR r, lanes 0-15 -> M=r ; lanes 16-31 -> M=8+r ; N=lane&15
  int col  = r0 + (lane & 15);
  int trow = t0 + ((lane >> 4) << 3);
  float bb = bias[col];
#pragma unroll
  for (int r = 0; r < 8; ++r)
    out[(size_t)(trow + r) * (size_t)R + col] = acc[r] + bb;
}

// Fused LSTM step: gates = [Wih@x] + Whh@h + (precomp-with-bias | bias),
// then elementwise cell update. One WG (8 waves) owns 32 hidden units:
// wave (gate, half) computes 16 gate rows via WMMA mat-vec.
// x/h vectors are async-staged into LDS once per WG (ASYNCcnt path), then
// all waves feed WMMA B operands from LDS.
__global__ void __launch_bounds__(256)
lstm_step_kernel(const u16* __restrict__ Wih, int xdim,
                 const u16* __restrict__ xbf,
                 const u16* __restrict__ Whh, int hdim,
                 const u16* __restrict__ hbf_in,
                 const float* __restrict__ precomp,  // row t, incl. bias; or null
                 const float* __restrict__ bias,
                 int H,
                 float* __restrict__ c_state,
                 float* __restrict__ h_out_f,
                 u16*   __restrict__ h_out_bf,
                 u16*   __restrict__ h_out_bf2,   // optional (ys0 row)
                 float* __restrict__ h_out_f2) {  // optional (decoded row)
  __shared__ __align__(16) u16 xs[IN];
  __shared__ __align__(16) u16 hs[HID];
  __shared__ float lds_g[4][32];
  int lane  = threadIdx.x & 31;
  int wave  = threadIdx.x >> 5;
  int gate  = wave >> 1;
  int half  = wave & 1;
  int unit0 = blockIdx.x * 32;
  int row0  = gate * H + unit0 + half * 16;

  // ---- async stage of the mat-vec B vectors into LDS (wave 0) ----
  if (wave == 0) {
    if (Wih) async_stage_vec(xbf, (u32)(uintptr_t)xs, xdim, lane);
    async_stage_vec(hbf_in, (u32)(uintptr_t)hs, hdim, lane);
    asm volatile("s_wait_asynccnt 0x0" ::: "memory");
  }
  __syncthreads();

  v8f acc = {};
  if (Wih) {
    for (int k = 0; k < xdim; k += 32) {
      if (k + 32 < xdim)
        __builtin_prefetch(Wih + (size_t)(row0 + (lane & 15)) * xdim + k + 32, 0, 1);
      v16bf a = load_a_tile(Wih, xdim, row0, k, lane);
      v16bf b = load_b_vec(xs, k, lane);
      acc = __builtin_amdgcn_wmma_f32_16x16x32_bf16(false, a, false, b,
                                                    (short)0, acc, false, false);
    }
  }
  for (int k = 0; k < hdim; k += 32) {
    if (k + 32 < hdim)
      __builtin_prefetch(Whh + (size_t)(row0 + (lane & 15)) * hdim + k + 32, 0, 1);
    v16bf a = load_a_tile(Whh, hdim, row0, k, lane);
    v16bf b = load_b_vec(hs, k, lane);
    acc = __builtin_amdgcn_wmma_f32_16x16x32_bf16(false, a, false, b,
                                                  (short)0, acc, false, false);
  }

  // All 16 D-columns are identical (B replicated); lanes 0/16 hold the result.
  if ((lane & 15) == 0) {
    int o = half * 16 + ((lane >> 4) << 3);
#pragma unroll
    for (int r = 0; r < 8; ++r) lds_g[gate][o + r] = acc[r];
  }
  __syncthreads();

  if (threadIdx.x < 32) {
    int u = unit0 + threadIdx.x;
    float bi, bf_, bg, bo;
    if (precomp) {
      bi = precomp[0 * H + u]; bf_ = precomp[1 * H + u];
      bg = precomp[2 * H + u]; bo  = precomp[3 * H + u];
    } else {
      bi = bias[0 * H + u]; bf_ = bias[1 * H + u];
      bg = bias[2 * H + u]; bo  = bias[3 * H + u];
    }
    float gi = lds_g[0][threadIdx.x] + bi;
    float gf = lds_g[1][threadIdx.x] + bf_;
    float gg = lds_g[2][threadIdx.x] + bg;
    float go = lds_g[3][threadIdx.x] + bo;
    float cn = sigm(gf) * c_state[u] + sigm(gi) * tanhf(gg);
    float hn = sigm(go) * tanhf(cn);
    c_state[u] = cn;
    h_out_f[u] = hn;
    u16 hb = f2bf(hn);
    h_out_bf[u] = hb;
    if (h_out_bf2) h_out_bf2[u] = hb;
    if (h_out_f2)  h_out_f2[u]  = hn;
  }
}

// Decoder output projection: x_new(bf16) = W(rows x K)@h + bias
__global__ void __launch_bounds__(256)
matvec_bf_kernel(const u16* __restrict__ W, int K,
                 const u16* __restrict__ x, const float* __restrict__ bias,
                 u16* __restrict__ out_bf) {
  int lane = threadIdx.x & 31;
  int wave = threadIdx.x >> 5;
  int row0 = (blockIdx.x * 8 + wave) * 16;
  v8f acc = {};
  for (int k = 0; k < K; k += 32) {
    v16bf a = load_a_tile(W, K, row0, k, lane);
    v16bf b = load_b_vec(x, k, lane);
    acc = __builtin_amdgcn_wmma_f32_16x16x32_bf16(false, a, false, b,
                                                  (short)0, acc, false, false);
  }
  if ((lane & 15) == 0) {
    int base = row0 + ((lane >> 4) << 3);
#pragma unroll
    for (int r = 0; r < 8; ++r)
      out_bf[base + r] = f2bf(acc[r] + bias[base + r]);
  }
}

// Reparameterization + decoder state init + d_out mu/logvar
__global__ void __launch_bounds__(256)
vae_prep_kernel(const float* __restrict__ h0T, const float* __restrict__ c0T,
                const float* __restrict__ h1T, const float* __restrict__ c1T,
                const float* __restrict__ eps,
                float* __restrict__ out_mu, float* __restrict__ out_logvar,
                float* __restrict__ dc0, float* __restrict__ dc1,
                u16* __restrict__ dh0_bf, u16* __restrict__ dh1_bf,
                u16* __restrict__ x_bf) {
  int i = blockIdx.x * 256 + threadIdx.x;
  if (i < IN) x_bf[i] = 0;   // x0 = zeros
  if (i < DH) {
    float mu0 = h0T[i], lv0 = h0T[i + DH];
    float z0 = mu0 + eps[i] * __expf(0.5f * lv0);
    out_mu[i] = mu0; out_logvar[i] = lv0;
    dh0_bf[i] = f2bf(z0); dc0[i] = c0T[i];
    float mu1 = h1T[i], lv1 = h1T[i + DH];
    float z1 = mu1 + eps[DH + i] * __expf(0.5f * lv1);
    out_mu[DH + i] = mu1; out_logvar[DH + i] = lv1;
    dh1_bf[i] = f2bf(z1); dc1[i] = c1T[i];
  }
}

// ---------------------------------------------------------------- host

extern "C" void kernel_launch(void* const* d_in, const int* in_sizes, int n_in,
                              void* d_out, int out_size, void* d_ws, size_t ws_size,
                              hipStream_t stream) {
  const float* features = (const float*)d_in[0];
  const float* eps      = (const float*)d_in[1];
  const float* eW_ih0   = (const float*)d_in[2];
  const float* eW_hh0   = (const float*)d_in[3];
  const float* eb0      = (const float*)d_in[4];
  const float* eW_ih1   = (const float*)d_in[5];
  const float* eW_hh1   = (const float*)d_in[6];
  const float* eb1      = (const float*)d_in[7];
  const float* dW_ih0   = (const float*)d_in[8];
  const float* dW_hh0   = (const float*)d_in[9];
  const float* db0      = (const float*)d_in[10];
  const float* dW_ih1   = (const float*)d_in[11];
  const float* dW_hh1   = (const float*)d_in[12];
  const float* db1      = (const float*)d_in[13];
  const float* outW     = (const float*)d_in[14];
  const float* outb     = (const float*)d_in[15];

  float* out_mu     = (float*)d_out;            // (2,1,1024)
  float* out_logvar = out_mu + 2 * DH;          // (2,1,1024)
  float* out_dec    = out_logvar + 2 * DH;      // (1024,1,1024), time-reversed

  // ---- workspace carve (256B aligned) ----
  char* ws = (char*)d_ws;
  size_t off = 0;
  auto alloc = [&](size_t bytes) -> void* {
    off = (off + 255) & ~(size_t)255;
    void* p = ws + off;
    off += bytes;
    return p;
  };
  u16* eWih0_bf = (u16*)alloc((size_t)4 * HID * IN  * 2);
  u16* eWhh0_bf = (u16*)alloc((size_t)4 * HID * HID * 2);
  u16* eWih1_bf = (u16*)alloc((size_t)4 * HID * HID * 2);
  u16* eWhh1_bf = (u16*)alloc((size_t)4 * HID * HID * 2);
  u16* dWih0_bf = (u16*)alloc((size_t)4 * DH * IN * 2);
  u16* dWhh0_bf = (u16*)alloc((size_t)4 * DH * DH * 2);
  u16* dWih1_bf = (u16*)alloc((size_t)4 * DH * DH * 2);
  u16* dWhh1_bf = (u16*)alloc((size_t)4 * DH * DH * 2);
  u16* outW_bf  = (u16*)alloc((size_t)IN * DH * 2);
  u16* feat_bf  = (u16*)alloc((size_t)S * IN * 2);
  u16* ys0_bf   = (u16*)alloc((size_t)S * HID * 2);
  float* precomp0 = (float*)alloc((size_t)S * 4 * HID * 4);
  float* precomp1 = (float*)alloc((size_t)S * 4 * HID * 4);
  float* h0_f = (float*)alloc(HID * 4);
  float* c0_f = (float*)alloc(HID * 4);
  float* h1_f = (float*)alloc(HID * 4);
  float* c1_f = (float*)alloc(HID * 4);
  u16* h0_bf  = (u16*)alloc(2 * HID * 2);   // ping-pong
  u16* h1_bf  = (u16*)alloc(2 * HID * 2);
  float* dh0_f = (float*)alloc(DH * 4);
  float* dh1_f = (float*)alloc(DH * 4);
  float* dc0   = (float*)alloc(DH * 4);
  float* dc1   = (float*)alloc(DH * 4);
  u16* dh0_bf  = (u16*)alloc(2 * DH * 2);   // ping-pong
  u16* dh1_bf  = (u16*)alloc(2 * DH * 2);
  u16* x_bf    = (u16*)alloc(IN * 2);
  (void)ws_size; (void)n_in; (void)in_sizes; (void)out_size;

  // ---- one-time bf16 weight/feature caches (then L2-resident) ----
  struct Cv { const float* src; u16* dst; int n; };
  const Cv cvs[] = {
    { eW_ih0, eWih0_bf, 4 * HID * IN  }, { eW_hh0, eWhh0_bf, 4 * HID * HID },
    { eW_ih1, eWih1_bf, 4 * HID * HID }, { eW_hh1, eWhh1_bf, 4 * HID * HID },
    { dW_ih0, dWih0_bf, 4 * DH * IN },   { dW_hh0, dWhh0_bf, 4 * DH * DH },
    { dW_ih1, dWih1_bf, 4 * DH * DH },   { dW_hh1, dWhh1_bf, 4 * DH * DH },
    { outW,   outW_bf,  IN * DH },       { features, feat_bf, S * IN },
  };
  for (const Cv& c : cvs)
    cvt_bf16_kernel<<<(c.n + 255) / 256, 256, 0, stream>>>(c.src, c.dst, c.n);

  // ---- encoder layer 0 ----
  // input projection as dense WMMA GEMM (bias folded in)
  gemm_proj_kernel<<<dim3(4 * HID / 128, S / 16), 256, 0, stream>>>(
      feat_bf, eWih0_bf, eb0, precomp0, IN, 4 * HID);
  zero_state_kernel<<<HID / 256, 256, 0, stream>>>(h0_f, c0_f, h0_bf, HID);
  for (int t = 0; t < S; ++t) {
    int rp = t & 1, wp = 1 - rp;
    lstm_step_kernel<<<HID / 32, 256, 0, stream>>>(
        nullptr, 0, nullptr,
        eWhh0_bf, HID, h0_bf + rp * HID,
        precomp0 + (size_t)t * 4 * HID, eb0, HID,
        c0_f, h0_f, h0_bf + wp * HID,
        ys0_bf + (size_t)t * HID, nullptr);
  }

  // ---- encoder layer 1 ----
  gemm_proj_kernel<<<dim3(4 * HID / 128, S / 16), 256, 0, stream>>>(
      ys0_bf, eWih1_bf, eb1, precomp1, HID, 4 * HID);
  zero_state_kernel<<<HID / 256, 256, 0, stream>>>(h1_f, c1_f, h1_bf, HID);
  for (int t = 0; t < S; ++t) {
    int rp = t & 1, wp = 1 - rp;
    lstm_step_kernel<<<HID / 32, 256, 0, stream>>>(
        nullptr, 0, nullptr,
        eWhh1_bf, HID, h1_bf + rp * HID,
        precomp1 + (size_t)t * 4 * HID, eb1, HID,
        c1_f, h1_f, h1_bf + wp * HID,
        nullptr, nullptr);
  }

  // ---- reparameterize + decoder init (writes mu/logvar to d_out) ----
  vae_prep_kernel<<<IN / 256, 256, 0, stream>>>(
      h0_f, c0_f, h1_f, c1_f, eps,
      out_mu, out_logvar, dc0, dc1, dh0_bf, dh1_bf, x_bf);

  // ---- decoder: 1024 sequential steps of (cell0 -> cell1 -> out-proj) ----
  for (int t = 0; t < S; ++t) {
    int rp = t & 1, wp = 1 - rp;
    lstm_step_kernel<<<DH / 32, 256, 0, stream>>>(
        dWih0_bf, IN, x_bf,
        dWhh0_bf, DH, dh0_bf + rp * DH,
        nullptr, db0, DH,
        dc0, dh0_f, dh0_bf + wp * DH, nullptr, nullptr);
    lstm_step_kernel<<<DH / 32, 256, 0, stream>>>(
        dWih1_bf, DH, dh0_bf + wp * DH,
        dWhh1_bf, DH, dh1_bf + rp * DH,
        nullptr, db1, DH,
        dc1, dh1_f, dh1_bf + wp * DH, nullptr,
        out_dec + (size_t)(S - 1 - t) * DH);   // decoded_features = dec_hs[::-1]
    if (t != S - 1)
      matvec_bf_kernel<<<IN / 128, 256, 0, stream>>>(
          outW_bf, DH, dh1_bf + wp * DH, outb, x_bf);
  }
}
